// MultiHeadAttention_79714593014244
// MI455X (gfx1250) — matrix operand order repaired and checked by reading, not compiled, not run on hardware
//
#include <hip/hip_runtime.h>

// ---------------------------------------------------------------------------
// Problem constants (from the reference)
// ---------------------------------------------------------------------------
#define DIM   4096
#define NTOK  8192
#define NH    32
#define HD    128

typedef __attribute__((ext_vector_type(16))) __bf16 v16bf;
typedef __attribute__((ext_vector_type(8)))  float  v8f;

// ---------------------------------------------------------------------------
// bf16 <-> f32 helpers (round-to-nearest-even)
// ---------------------------------------------------------------------------
__device__ __forceinline__ unsigned short f32_to_bf16(float f) {
    unsigned int u = __float_as_uint(f);
    unsigned int r = u + 0x7FFFu + ((u >> 16) & 1u);
    return (unsigned short)(r >> 16);
}
__device__ __forceinline__ float bf16_to_f32(unsigned short h) {
    return __uint_as_float(((unsigned int)h) << 16);
}

// ---------------------------------------------------------------------------
// f32 -> bf16 conversion kernel (vectorized x4) for x
// ---------------------------------------------------------------------------
__global__ void convert_f32_to_bf16(const float* __restrict__ src,
                                    unsigned short* __restrict__ dst, int n) {
    int i = (blockIdx.x * blockDim.x + threadIdx.x) * 4;
    if (i + 3 < n) {
        float4 v = *reinterpret_cast<const float4*>(src + i);
        ushort4 o;
        o.x = f32_to_bf16(v.x);
        o.y = f32_to_bf16(v.y);
        o.z = f32_to_bf16(v.z);
        o.w = f32_to_bf16(v.w);
        *reinterpret_cast<ushort4*>(dst + i) = o;
    }
}

// ---------------------------------------------------------------------------
// Transpose + convert: W[K][N] f32  ->  Wt[N][K] bf16  (tiled via LDS)
// Pays the transpose once so the GEMM never needs LDS transpose stores.
// ---------------------------------------------------------------------------
__global__ void __launch_bounds__(256)
transpose_convert_bf16(const float* __restrict__ src, unsigned short* __restrict__ dst,
                       int K, int N) {
    __shared__ float tile[32][33];
    const int k0 = blockIdx.y * 32;
    const int n0 = blockIdx.x * 32;
    const int tx = threadIdx.x & 31;
    const int ty = threadIdx.x >> 5;  // 0..7
#pragma unroll
    for (int r = ty; r < 32; r += 8)
        tile[r][tx] = src[(size_t)(k0 + r) * N + (n0 + tx)];
    __syncthreads();
#pragma unroll
    for (int r = ty; r < 32; r += 8)
        dst[(size_t)(n0 + r) * K + (k0 + tx)] = f32_to_bf16(tile[tx][r]);
}

// ---------------------------------------------------------------------------
// Tiled bf16 GEMM with v_wmma_f32_16x16x32_bf16, double-buffered LDS.
//   C[M,N] = A[M,K] * B[K,N]  where B is supplied PRE-TRANSPOSED: Bt[N][K].
// Block tile 256x128, BK = 32; 256 threads = 8 waves, each wave owns a
// 64x64 patch = 4x4 tiles of 16x16 (128 accumulator VGPRs).
// Per k-step per wave: 4 A-frags + 4 B-frags (16 ds_load_b128) : 16 WMMA.
//
// Fragment layouts per CDNA5 ISA §7.12.2 (wave32):
//   A 16x32 bf16 : lane L (half=L>>4, m=L&15): elems 0..7 = K[half*8 .. +7],
//                  elems 8..15 = K[16+half*8 .. +7]
//   B 32x16 bf16 : lane L: col n=L&15, elems 0..7 = K[half*8 .. +7],
//                  elems 8..15 = K[16+half*8 .. +7] (same order as A)
//   C/D 16x16 f32: elem r -> row = half*8 + r, col = lane&15.
// ---------------------------------------------------------------------------
template <bool OUT_BF16>
__global__ void __launch_bounds__(256)
gemm_wmma_bf16(const unsigned short* __restrict__ A,   // [M][K] bf16
               const unsigned short* __restrict__ Bt,  // [N][K] bf16 (transposed)
               void* __restrict__ Cout,
               int M, int N, int K) {
    constexpr int BM = 256, BN = 128, BK = 32;
    constexpr int LDT = 40;  // padded row stride (bf16) -> 80 B, 16B-aligned rows

    __shared__ unsigned short As[2][BM * LDT];  // [buf][row][k]   2 x 20 KB
    __shared__ unsigned short Bs[2][BN * LDT];  // [buf][n][k]     2 x 10 KB

    const int tid  = threadIdx.x;
    const int lane = tid & 31;
    const int wave = tid >> 5;
    const int half = (lane >> 4) & 1;
    const int l16  = lane & 15;

    const int wr = wave >> 1;  // 0..3 : M offset wr*64
    const int wc = wave & 1;   // 0..1 : N offset wc*64

    const int blockM = blockIdx.y * BM;
    const int blockN = blockIdx.x * BN;

    v8f acc[4][4];
#pragma unroll
    for (int i = 0; i < 4; ++i)
#pragma unroll
        for (int j = 0; j < 4; ++j)
            acc[i][j] = (v8f){0.f, 0.f, 0.f, 0.f, 0.f, 0.f, 0.f, 0.f};

    // staging: each thread copies one 16-bf16 row segment (2 x uint4)
    const unsigned short* aptr = A  + (size_t)(blockM + tid) * K;          // all 256 rows
    const unsigned short* bptr = Bt + (size_t)(blockN + (tid & 127)) * K;  // 128 rows

    union Frag { uint4 u[2]; v16bf v; };

    const int NSTEPS = K / BK;

    // ---- prologue: stage tile 0 into buffer 0 ----
    {
        uint4 a0 = reinterpret_cast<const uint4*>(aptr)[0];
        uint4 a1 = reinterpret_cast<const uint4*>(aptr)[1];
        *reinterpret_cast<uint4*>(&As[0][tid * LDT])     = a0;
        *reinterpret_cast<uint4*>(&As[0][tid * LDT + 8]) = a1;
        if (tid < BN) {
            uint4 b0 = reinterpret_cast<const uint4*>(bptr)[0];
            uint4 b1 = reinterpret_cast<const uint4*>(bptr)[1];
            *reinterpret_cast<uint4*>(&Bs[0][tid * LDT])     = b0;
            *reinterpret_cast<uint4*>(&Bs[0][tid * LDT + 8]) = b1;
        }
    }
    __syncthreads();

    for (int kk = 0; kk < NSTEPS; ++kk) {
        const int cur = kk & 1;
        const int nxt = cur ^ 1;
        const bool more = (kk + 1) < NSTEPS;

        // ---- prefetch next tile into registers (latency overlapped) ----
        uint4 pa0, pa1, pb0, pb1;
        if (more) {
            const unsigned short* ap = aptr + (kk + 1) * BK;
            pa0 = reinterpret_cast<const uint4*>(ap)[0];
            pa1 = reinterpret_cast<const uint4*>(ap)[1];
            if (tid < BN) {
                const unsigned short* bp = bptr + (kk + 1) * BK;
                pb0 = reinterpret_cast<const uint4*>(bp)[0];
                pb1 = reinterpret_cast<const uint4*>(bp)[1];
            }
        }

        // ---- fragments from LDS (contiguous 16B chunks) ----
        v16bf afrag[4];
#pragma unroll
        for (int i = 0; i < 4; ++i) {
            const int m = wr * 64 + i * 16 + l16;
            Frag f;
            f.u[0] = *reinterpret_cast<const uint4*>(&As[cur][m * LDT + half * 8]);
            f.u[1] = *reinterpret_cast<const uint4*>(&As[cur][m * LDT + 16 + half * 8]);
            afrag[i] = f.v;
        }
        v16bf bfrag[4];
#pragma unroll
        for (int j = 0; j < 4; ++j) {
            const int n = wc * 64 + j * 16 + l16;
            Frag f;
            f.u[0] = *reinterpret_cast<const uint4*>(&Bs[cur][n * LDT + half * 8]);
            f.u[1] = *reinterpret_cast<const uint4*>(&Bs[cur][n * LDT + 16 + half * 8]);
            bfrag[j] = f.v;
        }

#pragma unroll
        for (int i = 0; i < 4; ++i)
#pragma unroll
            for (int j = 0; j < 4; ++j)
                acc[i][j] = __builtin_amdgcn_wmma_f32_16x16x32_bf16(
                    false, afrag[i], false, bfrag[j],
                    (short)0, acc[i][j], false, false);

        // ---- store prefetch into the other buffer ----
        if (more) {
            *reinterpret_cast<uint4*>(&As[nxt][tid * LDT])     = pa0;
            *reinterpret_cast<uint4*>(&As[nxt][tid * LDT + 8]) = pa1;
            if (tid < BN) {
                *reinterpret_cast<uint4*>(&Bs[nxt][tid * LDT])     = pb0;
                *reinterpret_cast<uint4*>(&Bs[nxt][tid * LDT + 8]) = pb1;
            }
        }
        __syncthreads();
    }

    // ---- write back: acc elem r -> row half*8+r, col l16 ----
#pragma unroll
    for (int i = 0; i < 4; ++i) {
#pragma unroll
        for (int j = 0; j < 4; ++j) {
            const int row0 = blockM + wr * 64 + i * 16 + half * 8;
            const int col  = blockN + wc * 64 + j * 16 + l16;
#pragma unroll
            for (int r = 0; r < 8; ++r) {
                const float val = acc[i][j][r];
                const size_t off = (size_t)(row0 + r) * N + col;
                if (OUT_BF16)
                    reinterpret_cast<unsigned short*>(Cout)[off] = f32_to_bf16(val);
                else
                    reinterpret_cast<float*>(Cout)[off] = val;
            }
        }
    }
}

// ---------------------------------------------------------------------------
// Per-token head-mixing attention: one 128-thread block per token.
//   S = Q_n K_n^T  (32x32), softmax rows, y = S V_n  (32x128)
// Yb may alias Qb: Q is fully staged to LDS before any write (barriered).
// ---------------------------------------------------------------------------
__global__ void __launch_bounds__(128)
attention_per_token(const unsigned short* Qb,
                    const unsigned short* Kb,
                    const unsigned short* Vb,
                    unsigned short* Yb) {
    __shared__ float Qs[NH][HD];
    __shared__ float Ks[NH][HD];
    __shared__ float Vs[NH][HD];
    __shared__ float S[NH][NH + 1];

    const size_t base = (size_t)blockIdx.x * DIM;

    for (int idx = threadIdx.x; idx < NH * HD; idx += blockDim.x) {
        const int h = idx >> 7, d = idx & 127;
        Qs[h][d] = bf16_to_f32(Qb[base + idx]);
        Ks[h][d] = bf16_to_f32(Kb[base + idx]);
        Vs[h][d] = bf16_to_f32(Vb[base + idx]);
    }
    __syncthreads();

    for (int p = threadIdx.x; p < NH * NH; p += blockDim.x) {
        const int h = p >> 5, g = p & 31;
        float s = 0.f;
#pragma unroll 8
        for (int d = 0; d < HD; ++d) s += Qs[h][d] * Ks[g][d];
        S[h][g] = s;
    }
    __syncthreads();

    if (threadIdx.x < NH) {
        const int h = threadIdx.x;
        float m = -1e30f;
#pragma unroll
        for (int g = 0; g < NH; ++g) m = fmaxf(m, S[h][g]);
        float sum = 0.f;
#pragma unroll
        for (int g = 0; g < NH; ++g) {
            const float e = __expf(S[h][g] - m);
            S[h][g] = e;
            sum += e;
        }
        const float inv = 1.f / sum;
#pragma unroll
        for (int g = 0; g < NH; ++g) S[h][g] *= inv;
    }
    __syncthreads();

    for (int idx = threadIdx.x; idx < NH * HD; idx += blockDim.x) {
        const int h = idx >> 7, d = idx & 127;
        float a = 0.f;
#pragma unroll
        for (int g = 0; g < NH; ++g) a += S[h][g] * Vs[g][d];
        Yb[base + idx] = f32_to_bf16(a);
    }
}

// ---------------------------------------------------------------------------
// Launcher
// ---------------------------------------------------------------------------
extern "C" void kernel_launch(void* const* d_in, const int* in_sizes, int n_in,
                              void* d_out, int out_size, void* d_ws, size_t ws_size,
                              hipStream_t stream) {
    const float* x  = (const float*)d_in[0];
    const float* Wq = (const float*)d_in[1];
    const float* Wk = (const float*)d_in[2];
    const float* Wv = (const float*)d_in[3];
    const float* Wo = (const float*)d_in[4];

    const size_t XE = (size_t)NTOK * DIM;  // 33,554,432 elems
    const size_t WE = (size_t)DIM * DIM;   // 16,777,216 elems

    unsigned short* ws  = (unsigned short*)d_ws;
    unsigned short* xb  = ws;         // x  bf16             [NTOK][DIM]
    unsigned short* wqb = xb  + XE;   // W_q^T bf16          [DIM][DIM]
    unsigned short* wkb = wqb + WE;
    unsigned short* wvb = wkb + WE;
    unsigned short* wob = wvb + WE;
    unsigned short* Qb  = wob + WE;   // Q bf16 (reused for y after attention)
    unsigned short* Kb  = Qb  + XE;
    unsigned short* Vb  = Kb  + XE;
    // total: 4*XE + 4*WE elems * 2B = 384 MiB of workspace

    const int CT = 256;
    convert_f32_to_bf16<<<(int)(XE / 4 + CT - 1) / CT, CT, 0, stream>>>(x, xb, (int)XE);

    const dim3 tgrid(DIM / 32, DIM / 32);
    transpose_convert_bf16<<<tgrid, 256, 0, stream>>>(Wq, wqb, DIM, DIM);
    transpose_convert_bf16<<<tgrid, 256, 0, stream>>>(Wk, wkb, DIM, DIM);
    transpose_convert_bf16<<<tgrid, 256, 0, stream>>>(Wv, wvb, DIM, DIM);
    transpose_convert_bf16<<<tgrid, 256, 0, stream>>>(Wo, wob, DIM, DIM);

    const dim3 ggrid(DIM / 128, NTOK / 256);  // (N tiles, M tiles)
    const dim3 gblock(256);

    gemm_wmma_bf16<true><<<ggrid, gblock, 0, stream>>>(xb, wqb, Qb, NTOK, DIM, DIM);
    gemm_wmma_bf16<true><<<ggrid, gblock, 0, stream>>>(xb, wkb, Kb, NTOK, DIM, DIM);
    gemm_wmma_bf16<true><<<ggrid, gblock, 0, stream>>>(xb, wvb, Vb, NTOK, DIM, DIM);

    attention_per_token<<<NTOK, 128, 0, stream>>>(Qb, Kb, Vb, Qb);  // y -> Qb

    gemm_wmma_bf16<false><<<ggrid, gblock, 0, stream>>>(Qb, wob, d_out, NTOK, DIM, DIM);
}
